// EncoderLayer3_30056181138125
// MI455X (gfx1250) — matrix operand order repaired and checked by reading, not compiled, not run on hardware
//
#include <hip/hip_runtime.h>

typedef _Float16 f16;
typedef _Float16 v8h  __attribute__((ext_vector_type(8)));
typedef _Float16 v16h __attribute__((ext_vector_type(16)));
typedef float    v8f  __attribute__((ext_vector_type(8)));

#define BB  2
#define CCH 64
#define HHH 48
#define WWW 48
#define LL  2304
#define BL  4608      // B*L
#define NHH 8
#define DKK 64
#define DVV 128
#define DNN 128
#define NSS 16
#define CM  256       // C*MULT

// ---------------- WMMA fragment helpers (gfx1250 16x16x32 f16 layouts) ----------------

static __device__ __forceinline__ v16h load16(const f16* p) {
  v8h lo = *(const v8h*)p;
  v8h hi = *(const v8h*)(p + 8);
  v16h r;
#pragma unroll
  for (int i = 0; i < 8; i++) { r[i] = lo[i]; r[i + 8] = hi[i]; }
  return r;
}

// A-matrix 16x32 f16: lane holds row m=lane%16; lanes<16: K=0..7,16..23; lanes>=16: K=8..15,24..31
static __device__ __forceinline__ v16h load_a_frag(const f16* A, int lda, int row, int kb, int lane) {
  const f16* p = A + (size_t)row * lda + kb + ((lane < 16) ? 0 : 8);
  v8h lo = *(const v8h*)p;
  v8h hi = *(const v8h*)(p + 16);
  v16h r;
#pragma unroll
  for (int i = 0; i < 8; i++) { r[i] = lo[i]; r[i + 8] = hi[i]; }
  return r;
}

// B-matrix 32x16 f16 (row-major KxN): lane holds col n=lane%16; lanes<16: K=0..7,16..23; lanes>=16: K=8..15,24..31
static __device__ __forceinline__ v16h load_b_frag(const f16* B, int ldb, int kb, int nb, int lane) {
  v16h r;
#pragma unroll
  for (int i = 0; i < 16; i++) {
    int k = ((i < 8) ? i : i + 8) + ((lane < 16) ? 0 : 8);
    r[i] = B[(size_t)(kb + k) * ldb + nb + (lane & 15)];
  }
  return r;
}

// ---------------- Generic batched WMMA GEMM: C = A(MxK,f16) @ B(KxN,f16) ----------------
// wave computes 16 x (NT*16) tile; NT is compile-time so the K-loop is branch-free and
// accumulators stay pinned (no phi copies). K%32==0, M%16==0.
template<int NT, bool F16OUT>
__global__ void k_gemm(const f16* __restrict__ A, const f16* __restrict__ Bw,
                       void* __restrict__ Cp,
                       int M, int K, int lda, int ldb, int ldc,
                       long long sA, long long sB, int bMod, long long sC) {
  int lane  = threadIdx.x;
  int mtile = blockIdx.y * blockDim.y + threadIdx.y;
  if (mtile * 16 >= M) return;                      // wave-uniform
  int z = blockIdx.z;
  A  += (long long)z * sA;
  Bw += (long long)(z % bMod) * sB;
  int row = mtile * 16 + (lane & 15);
  int nb0 = blockIdx.x * (NT * 16);
  v8f acc[NT];
#pragma unroll
  for (int nt = 0; nt < NT; nt++) acc[nt] = (v8f){};
  for (int kb = 0; kb < K; kb += 32) {
    v16h a = load_a_frag(A, lda, row, kb, lane);
#pragma unroll
    for (int nt = 0; nt < NT; nt++) {
      v16h b = load_b_frag(Bw, ldb, kb, nb0 + nt * 16, lane);
      acc[nt] = __builtin_amdgcn_wmma_f32_16x16x32_f16(false, a, false, b,
                                                       (short)0, acc[nt], false, false);
    }
  }
  int r0 = mtile * 16 + ((lane >> 4) << 3);
  int cb = lane & 15;
  long long base = (long long)z * sC;
#pragma unroll
  for (int nt = 0; nt < NT; nt++) {
#pragma unroll
    for (int j = 0; j < 8; j++) {
      long long idx = base + (long long)(r0 + j) * ldc + nb0 + nt * 16 + cb;
      if (F16OUT) ((f16*)Cp)[idx]   = (f16)acc[nt][j];
      else        ((float*)Cp)[idx] = acc[nt][j];
    }
  }
}

// ---------------- Flash attention ----------------
// Block = 4 waves = 2 row-tiles x 2 dv-halves. Each wave: 16 q rows x 64 v dims.
// launch_bounds(128,1) lifts the VGPR cap so Q fragments stay resident (no
// per-iteration scratch reloads). Butterfly max issues all 8 row bpermutes
// per step before consuming, so they clause under a single dscnt wait; the
// softmax denominator is accumulated per-lane and reduced once at the end.
__global__ void __launch_bounds__(128, 1)
k_flash(const f16* __restrict__ qkv, const f16* __restrict__ kT,
        const float* __restrict__ logsnr, f16* __restrict__ oh) {
  int lane = threadIdx.x;
  int w    = threadIdx.y;
  int rowt = blockIdx.x * 2 + (w >> 1);
  int dvh  = w & 1;
  int h = blockIdx.y;
  int b = blockIdx.z;
  __shared__ float plds[4][16][33];
  int qbase = rowt * 16;
  int qrow  = b * LL + qbase + (lane & 15);
  v16h qa0 = load_a_frag(qkv + h * 64, 2048, qrow, 0,  lane);
  v16h qa1 = load_a_frag(qkv + h * 64, 2048, qrow, 32, lane);
#pragma unroll
  for (int i = 0; i < 16; i++) {                 // fold 1/sqrt(DK)=0.125 into Q (exact in f16)
    qa0[i] = qa0[i] * (f16)0.125f;
    qa1[i] = qa1[i] * (f16)0.125f;
  }
  v8f o[4] = {};
  float rmax[8], rsum[8];
#pragma unroll
  for (int j = 0; j < 8; j++) { rmax[j] = -1e30f; rsum[j] = 0.f; }
  int half = lane >> 4, col = lane & 15;
  const f16* kTb = kT + (size_t)((b * 8 + h) * 64) * LL;

  for (int kb = 0; kb < LL; kb += 32) {
    v8f s0 = {}, s1 = {};
    s0 = __builtin_amdgcn_wmma_f32_16x16x32_f16(false, qa0, false,
           load_b_frag(kTb, LL, 0, kb, lane), (short)0, s0, false, false);
    s0 = __builtin_amdgcn_wmma_f32_16x16x32_f16(false, qa1, false,
           load_b_frag(kTb, LL, 32, kb, lane), (short)0, s0, false, false);
    s1 = __builtin_amdgcn_wmma_f32_16x16x32_f16(false, qa0, false,
           load_b_frag(kTb, LL, 0, kb + 16, lane), (short)0, s1, false, false);
    s1 = __builtin_amdgcn_wmma_f32_16x16x32_f16(false, qa1, false,
           load_b_frag(kTb, LL, 32, kb + 16, lane), (short)0, s1, false, false);
    float bia0 = logsnr[b * LL + kb + col];
    float bia1 = logsnr[b * LL + kb + 16 + col];
    float bm[8], t[8];
#pragma unroll
    for (int j = 0; j < 8; j++) {
      s0[j] += bia0;
      s1[j] += bia1;
      bm[j] = fmaxf(s0[j], s1[j]);
    }
    // butterfly max across the 16-lane row group; issue all 8 bpermutes of a
    // step before consuming so they share one dscnt wait
#pragma unroll
    for (int j = 0; j < 8; j++) t[j] = __shfl_xor(bm[j], 1, 32);
#pragma unroll
    for (int j = 0; j < 8; j++) bm[j] = fmaxf(bm[j], t[j]);
#pragma unroll
    for (int j = 0; j < 8; j++) t[j] = __shfl_xor(bm[j], 2, 32);
#pragma unroll
    for (int j = 0; j < 8; j++) bm[j] = fmaxf(bm[j], t[j]);
#pragma unroll
    for (int j = 0; j < 8; j++) t[j] = __shfl_xor(bm[j], 4, 32);
#pragma unroll
    for (int j = 0; j < 8; j++) bm[j] = fmaxf(bm[j], t[j]);
#pragma unroll
    for (int j = 0; j < 8; j++) t[j] = __shfl_xor(bm[j], 8, 32);
#pragma unroll
    for (int j = 0; j < 8; j++) bm[j] = fmaxf(bm[j], t[j]);
    int rb = half * 8;
#pragma unroll
    for (int j = 0; j < 8; j++) {
      float nm = fmaxf(rmax[j], bm[j]);
      float sc = __expf(rmax[j] - nm);      // uniform within the 16-lane group
      rmax[j] = nm;
      float p0 = __expf(s0[j] - nm);
      float p1 = __expf(s1[j] - nm);
      rsum[j] = rsum[j] * sc + p0 + p1;     // per-lane partial; reduced at end
#pragma unroll
      for (int nt = 0; nt < 4; nt++) o[nt][j] *= sc;
      plds[w][rb + j][col]      = p0;
      plds[w][rb + j][16 + col] = p1;
    }
    // rebuild probs as A fragment (cross-lane via LDS; same-wave DS is in-order)
    v16h pa;
    int m = lane & 15;
#pragma unroll
    for (int j2 = 0; j2 < 16; j2++) {
      int kk = (lane < 16) ? ((j2 < 8) ? j2 : j2 + 8) : ((j2 < 8) ? j2 + 8 : j2 + 16);
      pa[j2] = (f16)plds[w][m][kk];
    }
    const f16* vrow = qkv + (size_t)(b * LL) * 2048 + 1024 + h * 128 + dvh * 64;
#pragma unroll
    for (int nt = 0; nt < 4; nt++) {
      v16h vb = load_b_frag(vrow, 2048, kb, nt * 16, lane);
      o[nt] = __builtin_amdgcn_wmma_f32_16x16x32_f16(false, pa, false, vb,
                                                     (short)0, o[nt], false, false);
    }
  }
  // final denominator: reduce per-lane partial sums across the 16-lane group
  {
    float t[8];
#pragma unroll
    for (int j = 0; j < 8; j++) t[j] = __shfl_xor(rsum[j], 1, 32);
#pragma unroll
    for (int j = 0; j < 8; j++) rsum[j] += t[j];
#pragma unroll
    for (int j = 0; j < 8; j++) t[j] = __shfl_xor(rsum[j], 2, 32);
#pragma unroll
    for (int j = 0; j < 8; j++) rsum[j] += t[j];
#pragma unroll
    for (int j = 0; j < 8; j++) t[j] = __shfl_xor(rsum[j], 4, 32);
#pragma unroll
    for (int j = 0; j < 8; j++) rsum[j] += t[j];
#pragma unroll
    for (int j = 0; j < 8; j++) t[j] = __shfl_xor(rsum[j], 8, 32);
#pragma unroll
    for (int j = 0; j < 8; j++) rsum[j] += t[j];
  }
  int r0 = half * 8;
#pragma unroll
  for (int j = 0; j < 8; j++) {
    float inv = 1.f / rsum[j];
    size_t orow = (size_t)(b * LL + qbase + r0 + j) * 1024 + h * 128 + dvh * 64;
#pragma unroll
    for (int nt = 0; nt < 4; nt++)
      oh[orow + nt * 16 + col] = (f16)(o[nt][j] * inv);
  }
}

// ---------------- packing / cast kernels ----------------
__global__ void k_pack_xtok(const float* __restrict__ enc, f16* __restrict__ xh) {
  int i = blockIdx.x * blockDim.x + threadIdx.x;
  if (i >= BL * CCH) return;
  int c = i % CCH; int t = i / CCH; int b = t / LL; int l = t % LL;
  xh[i] = (f16)enc[((size_t)b * CCH + c) * LL + l];
}
__global__ void k_cast_place(const float* __restrict__ src, f16* __restrict__ dst,
                             int M, int N, int ldd, int coff) {
  int i = blockIdx.x * blockDim.x + threadIdx.x;
  if (i >= M * N) return;
  int r = i / N, c = i % N;
  dst[(size_t)r * ldd + coff + c] = (f16)src[i];
}
__global__ void k_transpose_cast(const float* __restrict__ src, f16* __restrict__ dst, int R, int Cc) {
  int i = blockIdx.x * blockDim.x + threadIdx.x;
  if (i >= R * Cc) return;
  int r = i / Cc, c = i % Cc;
  dst[(size_t)c * R + r] = (f16)src[i];
}
__global__ void k_pack_xproj(const float* __restrict__ w, f16* __restrict__ dst) {
  int i = blockIdx.x * blockDim.x + threadIdx.x;
  if (i >= 4 * DNN * 48) return;
  int c = i % 48; int rest = i / 48; int d = rest % DNN; int k = rest / DNN;
  dst[i] = (c < 36) ? (f16)w[((size_t)k * 36 + c) * DNN + d] : (f16)0.f;
}
__global__ void k_logsnr(const float* __restrict__ snr, float* __restrict__ out) {
  int i = blockIdx.x * blockDim.x + threadIdx.x;
  if (i < BL) out[i] = logf(snr[i] + 1e-4f);
}
__global__ void k_prep_kT(const f16* __restrict__ qkv, f16* __restrict__ kT) {
  int i = blockIdx.x * blockDim.x + threadIdx.x;
  if (i >= BB * NHH * 64 * LL) return;
  int l = i % LL; int rest = i / LL; int dk = rest % 64; rest /= 64;
  int h = rest % NHH; int b = rest / NHH;
  kT[i] = qkv[((size_t)b * LL + l) * 2048 + 512 + h * 64 + dk];
}

// ---------------- LayerNorm kernels ----------------
__global__ void k_ln_attn(const float* __restrict__ fcout, const float* __restrict__ enc,
                          const float* __restrict__ w, const float* __restrict__ bs,
                          f16* __restrict__ out_h) {
  int t = blockIdx.x, c = threadIdx.x;
  int b = t / LL, l = t % LL;
  float v = fcout[(size_t)t * CCH + c] + enc[((size_t)b * CCH + c) * LL + l];
  __shared__ float red[CCH];
  red[c] = v; __syncthreads();
  for (int s = 32; s > 0; s >>= 1) { if (c < s) red[c] += red[c + s]; __syncthreads(); }
  float mean = red[0] * (1.f / CCH); __syncthreads();
  float d = v - mean; red[c] = d * d; __syncthreads();
  for (int s = 32; s > 0; s >>= 1) { if (c < s) red[c] += red[c + s]; __syncthreads(); }
  float rsd = rsqrtf(red[0] * (1.f / CCH) + 1e-5f);
  out_h[(size_t)t * CCH + c] = (f16)(d * rsd * w[c] + bs[c]);
}
__global__ void k_ln_ffn(const float* __restrict__ x2, const float* __restrict__ w,
                         const float* __restrict__ bs, f16* __restrict__ out_h) {
  int t = blockIdx.x, c = threadIdx.x;
  float v = x2[(size_t)t * CCH + c];
  __shared__ float red[CCH];
  red[c] = v; __syncthreads();
  for (int s = 32; s > 0; s >>= 1) { if (c < s) red[c] += red[c + s]; __syncthreads(); }
  float mean = red[0] * (1.f / CCH); __syncthreads();
  float d = v - mean; red[c] = d * d; __syncthreads();
  for (int s = 32; s > 0; s >>= 1) { if (c < s) red[c] += red[c + s]; __syncthreads(); }
  float rsd = rsqrtf(red[0] * (1.f / CCH) + 1e-5f);
  out_h[(size_t)t * CCH + c] = (f16)(d * rsd * w[c] + bs[c]);
}
__global__ void k_combine_ln(const float* __restrict__ ys, const float* __restrict__ xz,
                             const float* __restrict__ snr, const float* __restrict__ nw,
                             const float* __restrict__ nb_, f16* __restrict__ y_h) {
  int t = blockIdx.x, d = threadIdx.x;
  int b = t / LL, l = t % LL;
  int hh = l / WWW, ww = l % WWW;
  int lwh = ww * HHH + hh;
  size_t base = (size_t)b * 4 * DNN * LL;
  float v = ys[base + (size_t)(0 * DNN + d) * LL + l]
          + ys[base + (size_t)(2 * DNN + d) * LL + (LL - 1 - l)]
          + ys[base + (size_t)(1 * DNN + d) * LL + lwh]
          + ys[base + (size_t)(3 * DNN + d) * LL + (LL - 1 - lwh)];
  __shared__ float red[DNN];
  red[d] = v; __syncthreads();
  for (int s = 64; s > 0; s >>= 1) { if (d < s) red[d] += red[d + s]; __syncthreads(); }
  float mean = red[0] * (1.f / DNN); __syncthreads();
  float dd = v - mean; red[d] = dd * dd; __syncthreads();
  for (int s = 64; s > 0; s >>= 1) { if (d < s) red[d] += red[d + s]; __syncthreads(); }
  float rsd = rsqrtf(red[0] * (1.f / DNN) + 1e-5f);
  float z = xz[(size_t)t * 256 + 128 + d];
  float sil = z / (1.f + __expf(-z));
  y_h[(size_t)t * DNN + d] = (f16)((dd * rsd * nw[d] + nb_[d]) * sil * snr[t]);
}

// ---------------- conv / elementwise ----------------
__global__ void k_ssconv(const float* __restrict__ xz, const float* __restrict__ cw,
                         const float* __restrict__ cb, f16* __restrict__ xs_t) {
  int tid = blockIdx.x * blockDim.x + threadIdx.x;
  if (tid >= BB * LL * DNN) return;
  int d = tid % DNN; int rest = tid / DNN; int l = rest % LL; int b = rest / LL;
  int hh = l / WWW, ww = l % WWW;
  float s = cb[d];
#pragma unroll
  for (int dy = 0; dy < 3; dy++) {
    int yy = hh + dy - 1;
    if (yy < 0 || yy >= HHH) continue;
#pragma unroll
    for (int dx = 0; dx < 3; dx++) {
      int xx = ww + dx - 1;
      if (xx < 0 || xx >= WWW) continue;
      s += xz[((size_t)b * LL + yy * WWW + xx) * 256 + d] * cw[d * 9 + dy * 3 + dx];
    }
  }
  float v = s / (1.f + __expf(-s));        // silu
  f16 hv = (f16)v;
  size_t b4 = (size_t)b * 4 * LL * DNN;
  xs_t[b4 + 0 * (size_t)LL * DNN + (size_t)l * DNN + d] = hv;
  xs_t[b4 + 1 * (size_t)LL * DNN + (size_t)(ww * HHH + hh) * DNN + d] = hv;
  xs_t[b4 + 2 * (size_t)LL * DNN + (size_t)(LL - 1 - l) * DNN + d] = hv;
  xs_t[b4 + 3 * (size_t)LL * DNN + (size_t)(LL - 1 - (ww * HHH + hh)) * DNN + d] = hv;
}
__global__ void k_gelu(float* __restrict__ x, int n) {
  int i = blockIdx.x * blockDim.x + threadIdx.x;
  if (i < n) { float v = x[i]; x[i] = 0.5f * v * (1.f + erff(v * 0.70710678118f)); }
}
__global__ void k_dwconv(const float* __restrict__ h1, const float* __restrict__ dwc,
                         f16* __restrict__ h2) {
  int tid = blockIdx.x * blockDim.x + threadIdx.x;
  if (tid >= BB * LL * CM) return;
  int o = tid % CM; int rest = tid / CM; int l = rest % LL; int b = rest / LL;
  int hh = l / WWW, ww = l % WWW;
  float s = 0.f;
#pragma unroll
  for (int dy = 0; dy < 3; dy++) {
    int yy = hh + dy - 1;
    if (yy < 0 || yy >= HHH) continue;
#pragma unroll
    for (int dx = 0; dx < 3; dx++) {
      int xx = ww + dx - 1;
      if (xx < 0 || xx >= WWW) continue;
      s += h1[((size_t)b * LL + yy * WWW + xx) * CM + o] * dwc[o * 9 + dy * 3 + dx];
    }
  }
  h2[tid] = (f16)(0.5f * s * (1.f + erff(s * 0.70710678118f)));
}
__global__ void k_x2add(const float* __restrict__ g, const float* __restrict__ enc,
                        float* __restrict__ x2) {
  int i = blockIdx.x * blockDim.x + threadIdx.x;
  if (i >= BL * CCH) return;
  int c = i % CCH; int t = i / CCH; int b = t / LL; int l = t % LL;
  x2[i] = g[i] + enc[((size_t)b * CCH + c) * LL + l];
}
__global__ void k_final(const float* __restrict__ g, const float* __restrict__ x2,
                        float* __restrict__ out) {
  int i = blockIdx.x * blockDim.x + threadIdx.x;
  if (i >= BL * CCH) return;
  int c = i % CCH; int t = i / CCH; int b = t / LL; int l = t % LL;
  out[((size_t)b * CCH + c) * LL + l] = g[i] + x2[i];
}

// ---------------- selective scan ----------------
__global__ void k_scan(const float* __restrict__ xdblT, const f16* __restrict__ xs_t,
                       const float* __restrict__ dtw, const float* __restrict__ dtb,
                       const float* __restrict__ Alog, const float* __restrict__ Dp,
                       float* __restrict__ ys) {
  int k = blockIdx.x, b = blockIdx.y, d = threadIdx.x;
  float A[NSS];
#pragma unroll
  for (int n = 0; n < NSS; n++) A[n] = -__expf(Alog[((size_t)k * DNN + d) * NSS + n]);
  float w0 = dtw[((size_t)k * DNN + d) * 4 + 0];
  float w1 = dtw[((size_t)k * DNN + d) * 4 + 1];
  float w2 = dtw[((size_t)k * DNN + d) * 4 + 2];
  float w3 = dtw[((size_t)k * DNN + d) * 4 + 3];
  float bias = dtb[(size_t)k * DNN + d];
  float Dv = Dp[(size_t)k * DNN + d];
  float h[NSS];
#pragma unroll
  for (int n = 0; n < NSS; n++) h[n] = 0.f;
  __shared__ float srow[48];
  const float* xrow  = xdblT + ((size_t)(b * 4 + k)) * LL * 48;
  const f16*  xsrow  = xs_t  + ((size_t)(b * 4 + k)) * LL * DNN;
  float* yout = ys + ((size_t)((b * 4 + k) * DNN) + d) * LL;
  for (int l = 0; l < LL; l++) {
    __syncthreads();
    if (d < 48) srow[d] = xrow[(size_t)l * 48 + d];
    __syncthreads();
    float dtr = bias + w0 * srow[0] + w1 * srow[1] + w2 * srow[2] + w3 * srow[3];
    float dt = (dtr > 20.f) ? dtr : log1pf(__expf(dtr));
    float xt = (float)xsrow[(size_t)l * DNN + d];
    float dx = dt * xt;
    float y = 0.f;
#pragma unroll
    for (int n = 0; n < NSS; n++) {
      float dA = __expf(dt * A[n]);
      h[n] = h[n] * dA + dx * srow[4 + n];
      y += h[n] * srow[20 + n];
    }
    yout[l] = y + Dv * xt;
  }
}

// ---------------- host ----------------
extern "C" void kernel_launch(void* const* d_in, const int* in_sizes, int n_in,
                              void* d_out, int out_size, void* d_ws, size_t ws_size,
                              hipStream_t stream) {
  const float* enc_in     = (const float*)d_in[0];
  const float* snr        = (const float*)d_in[1];
  const float* wq         = (const float*)d_in[2];
  const float* wk         = (const float*)d_in[3];
  const float* wv         = (const float*)d_in[4];
  const float* fc         = (const float*)d_in[5];
  const float* aln_w      = (const float*)d_in[6];
  const float* aln_b      = (const float*)d_in[7];
  const float* ss_in_w    = (const float*)d_in[8];
  const float* ss_conv_w  = (const float*)d_in[9];
  const float* ss_conv_b  = (const float*)d_in[10];
  const float* ss_xproj_w = (const float*)d_in[11];
  const float* ss_dt_w    = (const float*)d_in[12];
  const float* ss_dt_b    = (const float*)d_in[13];
  const float* ss_Alog    = (const float*)d_in[14];
  const float* ss_D       = (const float*)d_in[15];
  const float* ss_norm_w  = (const float*)d_in[16];
  const float* ss_norm_b  = (const float*)d_in[17];
  const float* ss_out_w   = (const float*)d_in[18];
  const float* ffn_ln_w   = (const float*)d_in[19];
  const float* ffn_ln_b   = (const float*)d_in[20];
  const float* ffn_w1     = (const float*)d_in[21];
  const float* ffn_dw     = (const float*)d_in[22];
  const float* ffn_w2     = (const float*)d_in[23];
  float* out = (float*)d_out;

  char* ws = (char*)d_ws;
  size_t off = 0;
  auto alloc = [&](size_t bytes) -> char* {
    char* p = ws + off;
    off += (bytes + 255) & ~(size_t)255;
    return p;
  };
  f16*   x_h      = (f16*)  alloc((size_t)BL * CCH * 2);
  f16*   wqkv_h   = (f16*)  alloc((size_t)CCH * 2048 * 2);
  f16*   qkv_h    = (f16*)  alloc((size_t)BL * 2048 * 2);
  float* logsnr   = (float*)alloc((size_t)BL * 4);
  f16*   kT_h     = (f16*)  alloc((size_t)BB * NHH * 64 * LL * 2);
  f16*   o_h      = (f16*)  alloc((size_t)BL * 1024 * 2);
  f16*   fc_h     = (f16*)  alloc((size_t)1024 * CCH * 2);
  float* fcout    = (float*)alloc((size_t)BL * CCH * 4);
  f16*   enc_h    = (f16*)  alloc((size_t)BL * CCH * 2);
  f16*   ssin_h   = (f16*)  alloc((size_t)CCH * 256 * 2);
  float* xz       = (float*)alloc((size_t)BL * 256 * 4);
  f16*   xs_t     = (f16*)  alloc((size_t)BB * 4 * LL * DNN * 2);
  f16*   xprojT_h = (f16*)  alloc((size_t)4 * DNN * 48 * 2);
  float* xdblT    = (float*)alloc((size_t)BB * 4 * LL * 48 * 4);
  float* ysb      = (float*)alloc((size_t)BB * 4 * DNN * LL * 4);
  f16*   y_h      = (f16*)  alloc((size_t)BL * DNN * 2);
  f16*   outw_h   = (f16*)  alloc((size_t)DNN * CCH * 2);
  float* x2g      = (float*)alloc((size_t)BL * CCH * 4);
  float* x2       = (float*)alloc((size_t)BL * CCH * 4);
  f16*   xn_h     = (f16*)  alloc((size_t)BL * CCH * 2);
  f16*   w1t_h    = (f16*)  alloc((size_t)CCH * CM * 2);
  float* h1       = (float*)alloc((size_t)BL * CM * 4);
  f16*   w2t_h    = (f16*)  alloc((size_t)CM * CCH * 2);
  f16*   h2_h     = (f16*)  alloc((size_t)BL * CM * 2);
  float* f2out    = (float*)alloc((size_t)BL * CCH * 4);

  dim3 wblk(32, 4);
  auto eg = [](int n) { return dim3((n + 255) / 256); };

  // ---- pack weights / inputs ----
  k_pack_xtok<<<eg(BL * CCH), 256, 0, stream>>>(enc_in, x_h);
  k_cast_place<<<eg(64 * 512),  256, 0, stream>>>(wq, wqkv_h, 64, 512, 2048, 0);
  k_cast_place<<<eg(64 * 512),  256, 0, stream>>>(wk, wqkv_h, 64, 512, 2048, 512);
  k_cast_place<<<eg(64 * 1024), 256, 0, stream>>>(wv, wqkv_h, 64, 1024, 2048, 1024);
  k_cast_place<<<eg(1024 * 64), 256, 0, stream>>>(fc, fc_h, 1024, 64, 64, 0);
  k_cast_place<<<eg(64 * 256),  256, 0, stream>>>(ss_in_w, ssin_h, 64, 256, 256, 0);
  k_cast_place<<<eg(128 * 64),  256, 0, stream>>>(ss_out_w, outw_h, 128, 64, 64, 0);
  k_transpose_cast<<<eg(256 * 64), 256, 0, stream>>>(ffn_w1, w1t_h, 256, 64);
  k_transpose_cast<<<eg(64 * 256), 256, 0, stream>>>(ffn_w2, w2t_h, 64, 256);
  k_pack_xproj<<<eg(4 * 128 * 48), 256, 0, stream>>>(ss_xproj_w, xprojT_h);
  k_logsnr<<<eg(BL), 256, 0, stream>>>(snr, logsnr);

  // ---- attention ----
  k_gemm<4, true><<<dim3(32, 72, 1), wblk, 0, stream>>>(x_h, wqkv_h, qkv_h,
        BL, 64, 64, 2048, 2048, 0, 0, 1, 0);
  k_prep_kT<<<eg(BB * NHH * 64 * LL), 256, 0, stream>>>(qkv_h, kT_h);
  k_flash<<<dim3(72, NHH, BB), wblk, 0, stream>>>(qkv_h, kT_h, logsnr, o_h);
  k_gemm<4, false><<<dim3(1, 72, 1), wblk, 0, stream>>>(o_h, fc_h, fcout,
        BL, 1024, 1024, 64, 64, 0, 0, 1, 0);
  k_ln_attn<<<BL, CCH, 0, stream>>>(fcout, enc_in, aln_w, aln_b, enc_h);

  // ---- SS2D branch ----
  k_gemm<4, false><<<dim3(4, 72, 1), wblk, 0, stream>>>(enc_h, ssin_h, xz,
        BL, 64, 64, 256, 256, 0, 0, 1, 0);
  k_ssconv<<<eg(BB * LL * DNN), 256, 0, stream>>>(xz, ss_conv_w, ss_conv_b, xs_t);
  k_gemm<3, false><<<dim3(1, 36, 8), wblk, 0, stream>>>(xs_t, xprojT_h, xdblT,
        LL, 128, 128, 48, 48,
        (long long)LL * 128, (long long)128 * 48, 4, (long long)LL * 48);
  k_scan<<<dim3(4, BB), 128, 0, stream>>>(xdblT, xs_t, ss_dt_w, ss_dt_b, ss_Alog, ss_D, ysb);
  k_combine_ln<<<BL, DNN, 0, stream>>>(ysb, xz, snr, ss_norm_w, ss_norm_b, y_h);
  k_gemm<4, false><<<dim3(1, 72, 1), wblk, 0, stream>>>(y_h, outw_h, x2g,
        BL, 128, 128, 64, 64, 0, 0, 1, 0);
  k_x2add<<<eg(BL * CCH), 256, 0, stream>>>(x2g, enc_in, x2);

  // ---- FFN ----
  k_ln_ffn<<<BL, CCH, 0, stream>>>(x2, ffn_ln_w, ffn_ln_b, xn_h);
  k_gemm<4, false><<<dim3(4, 72, 1), wblk, 0, stream>>>(xn_h, w1t_h, h1,
        BL, 64, 64, 256, 256, 0, 0, 1, 0);
  k_gelu<<<eg(BL * CM), 256, 0, stream>>>(h1, BL * CM);
  k_dwconv<<<eg(BB * LL * CM), 256, 0, stream>>>(h1, ffn_dw, h2_h);
  k_gemm<4, false><<<dim3(1, 72, 1), wblk, 0, stream>>>(h2_h, w2t_h, f2out,
        BL, 256, 256, 64, 64, 0, 0, 1, 0);
  k_final<<<eg(BL * CCH), 256, 0, stream>>>(f2out, x2, out);
}
